// ALSGCN_76596446757443
// MI455X (gfx1250) — matrix operand order repaired and checked by reading, not compiled, not run on hardware
//
#include <hip/hip_runtime.h>
#include <cstdint>
#include <cstddef>

// ---------------------------------------------------------------------------
// ALSGCN block: x[N=64, C=64, T=300, V=17], K=3, groups=8
// Fused GEMM: M = C*V = 1088, Kdim = C*V = 1088, Ncol = N*T = 19200
// ---------------------------------------------------------------------------
typedef __attribute__((ext_vector_type(16))) __bf16 v16bf;
typedef __attribute__((ext_vector_type(8)))  float  v8f;

union V16U { v16bf v; uint4 q[2]; };

// ------------------------- workspace layout (bytes) ------------------------
constexpr size_t OFF_BW = 0;                          // bf16 [1088*1088]
constexpr size_t SZ_BW  = 1088ull * 1088 * 2;
constexpr size_t OFF_B2 = OFF_BW + SZ_BW;             // f32  [1088]
constexpr size_t SZ_B2  = 1088ull * 4;
constexpr size_t OFF_XB = OFF_B2 + SZ_B2;             // bf16 [19200*1088] col-major
constexpr size_t SZ_XB  = 19200ull * 1088 * 2;
constexpr size_t OFF_M2 = OFF_XB + SZ_XB;             // f32  [64*64*300*17]
constexpr size_t SZ_M2  = 20889600ull * 4;
constexpr size_t OFF_S1 = OFF_M2 + SZ_M2;             // f32  [4096]
constexpr size_t OFF_S2 = OFF_S1 + 16384;             // f32  [4096]
constexpr size_t OFF_AL = OFF_S2 + 16384;             // f32  [4096]
constexpr size_t OFF_BC = OFF_AL + 16384;             // f32  [64]

// ---------------------------------------------------------------------------
// Kernel 1: fold conv_w, A_base, GA, GMA*mask into one bf16 weight matrix
// ---------------------------------------------------------------------------
__global__ __launch_bounds__(256)
void alsgcn_prep_w(const float* __restrict__ A_base, const float* __restrict__ GA,
                   const float* __restrict__ GMA, const float* __restrict__ conv_w,
                   const float* __restrict__ conv_b,
                   __bf16* __restrict__ Bw, float* __restrict__ bias2)
{
    const int blk = blockIdx.x;            // 0..1087 == c*17 + w
    const int c = blk / 17, w = blk % 17, g = c & 7;
    __shared__ float Af[3][17];
    __shared__ float cw[3][64];
    const int tid = threadIdx.x;
    if (tid < 51) {
        int k = tid / 17, v = tid % 17;
        float mask = ((v >= 1 && v <= 8 && w >= 1 && w <= 8) ||
                      v == 13 || v == 14 || w == 13 || w == 14) ? 0.f : 1.f;
        Af[k][v] = 0.04f + A_base[(k * 17 + v) * 17 + w]
                 + GA[((k * 8 + g) * 17 + v) * 17 + w]
                 + GMA[((k * 8 + g) * 17 + v) * 17 + w] * mask;
    }
    if (tid >= 64 && tid < 256) {
        int t2 = tid - 64;
        int k = t2 / 64, cin = t2 % 64;
        cw[k][cin] = conv_w[(k * 64 + c) * 64 + cin];
    }
    __syncthreads();
    for (int r = tid; r < 1088; r += 256) {
        int cin = r / 17, v = r % 17;
        float s = cw[0][cin] * Af[0][v] + cw[1][cin] * Af[1][v] + cw[2][cin] * Af[2][v];
        Bw[(size_t)blk * 1088 + r] = (__bf16)s;
    }
    if (tid == 0) {
        float b = 0.f;
        for (int k = 0; k < 3; ++k) {
            float sv = 0.f;
            for (int v = 0; v < 17; ++v) sv += Af[k][v];
            b += conv_b[k * 64 + c] * sv;
        }
        bias2[blk] = b;
    }
}

// ---------------------------------------------------------------------------
// Kernel 2: x[n,cin,t,v] f32 -> Xb bf16 column-major: Xb[(n*300+t)*1088 + cin*17+v]
// ---------------------------------------------------------------------------
__global__ __launch_bounds__(256)
void alsgcn_prep_x(const float* __restrict__ x, __bf16* __restrict__ Xb)
{
    const int j = blockIdx.x;              // 0..19199 == n*300 + t
    const int n = j / 300, t = j % 300;
    const float* xb = x + (size_t)n * 64 * 5100 + (size_t)t * 17;
    __bf16* dst = Xb + (size_t)j * 1088;
    for (int i = threadIdx.x; i < 1088; i += 256) {
        int cin = i / 17, v = i % 17;
        dst[i] = (__bf16)xb[(size_t)cin * 5100 + v];
    }
}

// ---------------------------------------------------------------------------
// Async global->LDS 16B copy (gfx1250, ASYNCcnt-tracked, GVS addressing:
// per-lane LDS dest VGPR + per-lane u32 offset VGPR + uniform 64-bit SGPR base)
// ---------------------------------------------------------------------------
__device__ __forceinline__ void async_ld16(unsigned ldsaddr, unsigned voff,
                                           unsigned long long sbase)
{
    asm volatile("global_load_async_to_lds_b128 %0, %1, %2"
                 :: "v"(ldsaddr), "v"(voff), "s"(sbase) : "memory");
}
__device__ __forceinline__ void wait_async0()
{
    asm volatile("s_wait_asynccnt 0x0" ::: "memory");
}

// ---------------------------------------------------------------------------
// Kernel 3: fused GEMM. D[1088 x 19200] = Bw[1088x1088] * X[1088x19200]
// Block tile 64x256, 8 waves (2x4), wave tile 32x64 = 2x4 WMMA 16x16.
// BK=64 (two 32-K WMMA substeps per stage). Double-buffered LDS filled by
// GLOBAL_LOAD_ASYNC_TO_LDS_B128; pipeline: issue tile i+1 -> compute tile i
// -> s_wait_asynccnt 0 -> barrier.
// LDS pitch 144B: 16B aligned for ds b128; 36-bank row stride (conflict-free
// across the 16 row-lanes).
// Operand layouts per ISA 7.12.2:
//   A (16x32 bf16): lane(h=L/16,m=L%16): row m bytes [h*16,+16) and [32+h*16,+16)
//   B (32x16 bf16): lane: column L%16 bytes [h*16,+16) and [32+h*16,+16)
//   C/D: VGPR r -> M = r + 8*h, N = lane%16
// ---------------------------------------------------------------------------
__global__ __launch_bounds__(256)
void alsgcn_gemm(const __bf16* __restrict__ Bw, const __bf16* __restrict__ Xb,
                 const float* __restrict__ bias2, float* __restrict__ m2)
{
    constexpr int BM = 64, BN = 256, BK = 64;
    constexpr int PITCH = 144;                 // bytes per 64-K row (128 + 16 pad)
    constexpr int ASZ = BM * PITCH;            // 9216
    constexpr int XSZ = BN * PITCH;            // 36864
    constexpr int NIT = 1088 / BK;             // 17

    __shared__ __align__(16) unsigned char lds[2 * (ASZ + XSZ)];   // 92160 B

    const int tid  = threadIdx.x;
    const int lane = tid & 31;
    const int wave = tid >> 5;
    const int wr   = wave >> 2;            // 0..1 : 32-row strip
    const int wc   = wave & 3;             // 0..3 : 64-col strip
    const int h    = lane >> 4;
    const int l16  = lane & 15;
    const int m0   = blockIdx.y * BM;
    const int j0   = blockIdx.x * BN;

    v8f acc[2][4] = {};

    const unsigned char* gA = (const unsigned char*)Bw + (size_t)m0 * 2176;
    const unsigned char* gX = (const unsigned char*)Xb + (size_t)j0 * 2176;
    const unsigned ldsBase = (unsigned)(uintptr_t)(void*)lds;

    // per-thread chunk geometry (16B chunks): A tile = 512 chunks, X = 2048
    auto stage = [&](int buf, int kb) {
        const unsigned long long sA = (unsigned long long)(uintptr_t)gA + (unsigned)(kb * 2);
        const unsigned long long sX = (unsigned long long)(uintptr_t)gX + (unsigned)(kb * 2);
        const unsigned aB = ldsBase + buf * ASZ;
        const unsigned xB = ldsBase + 2 * ASZ + buf * XSZ;
#pragma unroll
        for (int i = 0; i < 2; ++i) {
            int c = tid + i * 256;
            int row = c >> 3, ch = c & 7;
            async_ld16(aB + row * PITCH + ch * 16, (unsigned)(row * 2176 + ch * 16), sA);
        }
#pragma unroll
        for (int u = 0; u < 8; ++u) {
            int c = tid + u * 256;
            int col = c >> 3, ch = c & 7;
            async_ld16(xB + col * PITCH + ch * 16, (unsigned)(col * 2176 + ch * 16), sX);
        }
    };

    stage(0, 0);
    wait_async0();
    __syncthreads();

    for (int it = 0; it < NIT; ++it) {
        const int cur = it & 1;
        if (it + 1 < NIT) stage(1 - cur, (it + 1) * BK);

        const unsigned char* cA = lds + cur * ASZ;
        const unsigned char* cX = lds + 2 * ASZ + cur * XSZ;
#pragma unroll
        for (int s = 0; s < 2; ++s) {          // two 32-K substeps
            V16U a[2], b[4];
#pragma unroll
            for (int i = 0; i < 2; ++i) {
                const unsigned char* p = cA + (wr * 32 + i * 16 + l16) * PITCH + s * 64 + h * 16;
                a[i].q[0] = *(const uint4*)(p);
                a[i].q[1] = *(const uint4*)(p + 32);
            }
#pragma unroll
            for (int jj = 0; jj < 4; ++jj) {
                const unsigned char* p = cX + (wc * 64 + jj * 16 + l16) * PITCH + s * 64 + h * 16;
                b[jj].q[0] = *(const uint4*)(p);
                b[jj].q[1] = *(const uint4*)(p + 32);
            }
#pragma unroll
            for (int i = 0; i < 2; ++i)
#pragma unroll
                for (int jj = 0; jj < 4; ++jj)
                    acc[i][jj] = __builtin_amdgcn_wmma_f32_16x16x32_bf16(
                        false, a[i].v, false, b[jj].v, (short)0, acc[i][jj],
                        false, false);
        }

        if (it + 1 < NIT) {
            wait_async0();
            __syncthreads();
        }
    }

    // Epilogue: add folded bias, scatter to m2[n,c,t,w] (x-compatible layout)
#pragma unroll
    for (int i = 0; i < 2; ++i) {
        const int mbase = m0 + wr * 32 + i * 16 + h * 8;
#pragma unroll
        for (int jj = 0; jj < 4; ++jj) {
            const int jg = j0 + wc * 64 + jj * 16 + l16;
            const int n = jg / 300, t = jg % 300;
#pragma unroll
            for (int r = 0; r < 8; ++r) {
                const int Mg = mbase + r;
                const int cch = Mg / 17, w = Mg % 17;
                float val = acc[i][jj][r] + bias2[Mg];
                __builtin_nontemporal_store(
                    val, &m2[(size_t)((n * 64 + cch) * 300 + t) * 17 + w]);
            }
        }
    }
}

// ---------------------------------------------------------------------------
// Kernel 4: per-(n,c) sum and sum-of-squares of m2 over (t,w)
// ---------------------------------------------------------------------------
__global__ __launch_bounds__(256)
void alsgcn_stats(const float* __restrict__ m2, float* __restrict__ S1,
                  float* __restrict__ S2)
{
    const int blk = blockIdx.x;            // n*64 + c
    const float* p = m2 + (size_t)blk * 5100;
    float s = 0.f, s2 = 0.f;
    for (int i = threadIdx.x; i < 5100; i += 256) {
        float v = p[i];
        s += v; s2 += v * v;
    }
    __shared__ float r1[256], r2[256];
    r1[threadIdx.x] = s; r2[threadIdx.x] = s2;
    __syncthreads();
    for (int off = 128; off > 0; off >>= 1) {
        if (threadIdx.x < off) {
            r1[threadIdx.x] += r1[threadIdx.x + off];
            r2[threadIdx.x] += r2[threadIdx.x + off];
        }
        __syncthreads();
    }
    if (threadIdx.x == 0) { S1[blk] = r1[0]; S2[blk] = r2[0]; }
}

// ---------------------------------------------------------------------------
// Kernel 5: ECA sigmoid gate + BatchNorm stats -> per-(n,c) alpha, per-c bias.
// q = m*(1+s[n,c]) so mu_c, E[q^2]_c follow in closed form from S1/S2.
// ---------------------------------------------------------------------------
__global__ __launch_bounds__(64)
void alsgcn_finalize(const float* __restrict__ S1, const float* __restrict__ S2,
                     const float* __restrict__ eca, const float* __restrict__ gamma,
                     const float* __restrict__ beta,
                     float* __restrict__ alpha, float* __restrict__ biasC)
{
    __shared__ float Q[4096];
    const int tid = threadIdx.x;
    for (int i = tid; i < 4096; i += 64) Q[i] = S1[i] * (1.f / 5100.f);
    __syncthreads();
    const float w0 = eca[0], w1 = eca[1], w2 = eca[2];
    const int c = tid;
    float smu = 0.f, sq2 = 0.f;
    for (int n = 0; n < 64; ++n) {
        float q  = Q[n * 64 + c];
        float qm = (c > 0)  ? Q[n * 64 + c - 1] : 0.f;
        float qp = (c < 63) ? Q[n * 64 + c + 1] : 0.f;
        float sg = 1.f / (1.f + expf(-(w0 * qm + w1 * q + w2 * qp)));
        float f  = 1.f + sg;
        smu += f * S1[n * 64 + c];
        sq2 += f * f * S2[n * 64 + c];
    }
    const float mu  = smu * (1.f / 326400.f);
    const float var = sq2 * (1.f / 326400.f) - mu * mu;
    const float inv = rsqrtf(var + 1e-5f);
    const float gi  = inv * gamma[c];
    biasC[c] = beta[c] - mu * gi;
    for (int n = 0; n < 64; ++n) {
        float q  = Q[n * 64 + c];
        float qm = (c > 0)  ? Q[n * 64 + c - 1] : 0.f;
        float qp = (c < 63) ? Q[n * 64 + c + 1] : 0.f;
        float sg = 1.f / (1.f + expf(-(w0 * qm + w1 * q + w2 * qp)));
        alpha[n * 64 + c] = (1.f + sg) * gi;
    }
}

// ---------------------------------------------------------------------------
// Kernel 6: out = relu(m2 * alpha[n,c] + biasC[c] + x)
// ---------------------------------------------------------------------------
__global__ __launch_bounds__(256)
void alsgcn_final(const float* __restrict__ m2, const float* __restrict__ x,
                  const float* __restrict__ alpha, const float* __restrict__ biasC,
                  float* __restrict__ out, int total)
{
    const int idx = blockIdx.x * 256 + threadIdx.x;
    if (idx >= total) return;
    const int nc = idx / 5100;             // == n*64 + c
    float v = __builtin_nontemporal_load(&m2[idx]) * alpha[nc]
            + biasC[nc & 63] + x[idx];
    out[idx] = v > 0.f ? v : 0.f;
}

// ---------------------------------------------------------------------------
extern "C" void kernel_launch(void* const* d_in, const int* in_sizes, int n_in,
                              void* d_out, int out_size, void* d_ws, size_t ws_size,
                              hipStream_t stream)
{
    (void)in_sizes; (void)n_in; (void)out_size; (void)ws_size;
    const float* x      = (const float*)d_in[0];
    const float* A_base = (const float*)d_in[1];
    const float* GA     = (const float*)d_in[2];
    const float* GMA    = (const float*)d_in[3];
    const float* conv_w = (const float*)d_in[4];
    const float* conv_b = (const float*)d_in[5];
    const float* eca    = (const float*)d_in[6];
    const float* gamma  = (const float*)d_in[7];
    const float* beta   = (const float*)d_in[8];
    float* out = (float*)d_out;

    char* ws = (char*)d_ws;
    __bf16* Bw    = (__bf16*)(ws + OFF_BW);
    float*  bias2 = (float*)(ws + OFF_B2);
    __bf16* Xb    = (__bf16*)(ws + OFF_XB);
    float*  m2    = (float*)(ws + OFF_M2);
    float*  S1    = (float*)(ws + OFF_S1);
    float*  S2    = (float*)(ws + OFF_S2);
    float*  alpha = (float*)(ws + OFF_AL);
    float*  biasC = (float*)(ws + OFF_BC);

    alsgcn_prep_w<<<1088, 256, 0, stream>>>(A_base, GA, GMA, conv_w, conv_b, Bw, bias2);
    alsgcn_prep_x<<<19200, 256, 0, stream>>>(x, Xb);
    alsgcn_gemm<<<dim3(75, 17), 256, 0, stream>>>(Bw, Xb, bias2, m2);
    alsgcn_stats<<<4096, 256, 0, stream>>>(m2, S1, S2);
    alsgcn_finalize<<<1, 64, 0, stream>>>(S1, S2, eca, gamma, beta, alpha, biasC);
    alsgcn_final<<<81600, 256, 0, stream>>>(m2, x, alpha, biasC, out, 20889600);
}
